// VisionEncoder_45758581571588
// MI455X (gfx1250) — compile-verified, matrix operand order checked by reading, not run on hardware
//
#include <hip/hip_runtime.h>

// ---------------------------------------------------------------------------
// VisionEncoder for MI455X (gfx1250, wave32, WMMA).
// - All GEMMs: f16 inputs / f32 accumulate via v_wmma_f32_16x16x32_f16.
// - GEMM tiles staged into LDS with GLOBAL_LOAD_ASYNC_TO_LDS_B128 (ASYNCcnt),
//   double-buffered: next K-tile loads overlap current K-tile WMMAs.
// - Residual stream x stays f32. LN / softmax in f32.
// ---------------------------------------------------------------------------

typedef __attribute__((ext_vector_type(16))) _Float16 v16h;
typedef __attribute__((ext_vector_type(8)))  float    v8f;
typedef __attribute__((ext_vector_type(4)))  int      v4i;

#define BDIM   256      // batch
#define HID    384
#define NP     196
#define NPAD   224      // 196 padded to multiple of 32 for attention K loop
#define FFND   1536
#define DOUTD  512
#define KPATCH 768      // 3*16*16
#define MROWS  50176    // 256*196 (== 392 * 128 exactly)

// ---------------- CDNA5 async global->LDS path ----------------

#if __has_builtin(__builtin_amdgcn_global_load_async_to_lds_b128)
#define HAVE_ASYNC_LDS 1
typedef __attribute__((address_space(1))) v4i* as1_v4i_ptr;
typedef __attribute__((address_space(3))) v4i* as3_v4i_ptr;
#endif

__device__ __forceinline__ void async_b128(const void* g, void* l) {
#ifdef HAVE_ASYNC_LDS
  __builtin_amdgcn_global_load_async_to_lds_b128(
      (as1_v4i_ptr)(v4i*)g, (as3_v4i_ptr)(v4i*)l, 0, 0);
#else
  *(uint4*)l = *(const uint4*)g;
#endif
}

template <int N>
__device__ __forceinline__ void wait_async() {
#ifdef HAVE_ASYNC_LDS
#if __has_builtin(__builtin_amdgcn_s_wait_asynccnt)
  __builtin_amdgcn_s_wait_asynccnt((short)N);
#else
  asm volatile("s_wait_asynccnt %0" ::"n"(N) : "memory");
#endif
#endif
}

// ---------------- WMMA helpers (layouts per CDNA5 ISA 7.12.2) ----------------

__device__ __forceinline__ v8f wmma_f16(v16h a, v16h b, v8f c) {
  // 8 args: (neg_a, A, neg_b, B, c_mod, C, reuse_a, reuse_b)
  return __builtin_amdgcn_wmma_f32_16x16x32_f16(false, a, false, b, (short)0, c,
                                                false, false);
}

// Load a 16x32 f16 fragment from row-major storage (row stride `ld` halfs).
// Works for A (rows = M) and for Bt (rows = N, i.e. B stored K-major per row).
// Lane L: row = L&15, half = L>>4; VGPR v holds K pair
//   k = (v&4 ? 16 : 0) + 8*half + 2*(v&3)   (half folded into base pointer)
__device__ __forceinline__ v16h frag_load(const _Float16* p, int ld) {
  int lane = threadIdx.x & 31;
  int r    = lane & 15;
  int h    = lane >> 4;
  const _Float16* row = p + (size_t)r * ld + h * 8;
  v16h f;
#pragma unroll
  for (int v = 0; v < 8; ++v) {
    int k = ((v & 4) ? 16 : 0) + (v & 3) * 2;
    f[2 * v]     = row[k];
    f[2 * v + 1] = row[k + 1];
  }
  return f;
}

// ---------------- elementwise / prep kernels ----------------

__global__ void f32_to_f16_kernel(const float* __restrict__ in,
                                  _Float16* __restrict__ out, long n) {
  long i = blockIdx.x * (long)blockDim.x + threadIdx.x;
  long st = (long)gridDim.x * blockDim.x;
  for (; i < n; i += st) out[i] = (_Float16)in[i];
}

// Concatenate Ws (rows 0..31) and Wa (rows 32..63) per layer: out [3][64][384]
__global__ void build_wsa_kernel(const float* __restrict__ Ws,
                                 const float* __restrict__ Wa,
                                 _Float16* __restrict__ out) {
  int i = blockIdx.x * blockDim.x + threadIdx.x;
  if (i >= 3 * 64 * HID) return;
  int l = i / (64 * HID);
  int r = (i / HID) % 64;
  int c = i % HID;
  float v = (r < 32) ? Ws[((size_t)l * 32 + r) * HID + c]
                     : Wa[((size_t)l * 32 + (r - 32)) * HID + c];
  out[i] = (_Float16)v;
}

// Zero the key-padding region of hT[b][c][196..224)
__global__ void zero_pad_hT_kernel(_Float16* __restrict__ hT) {
  int i = blockIdx.x * blockDim.x + threadIdx.x;
  const int total = BDIM * HID * (NPAD - NP);
  if (i >= total) return;
  int k  = NP + (i % (NPAD - NP));
  int bc = i / (NPAD - NP);
  hT[(size_t)bc * NPAD + k] = (_Float16)0.f;
}

// im2col: pixel[B,3,224,224] -> P16[B*196, 768] f16 (patch rows, conv-weight order)
__global__ void im2col_kernel(const float* __restrict__ px,
                              _Float16* __restrict__ out) {
  size_t i = blockIdx.x * (size_t)blockDim.x + threadIdx.x;
  const size_t total = (size_t)MROWS * KPATCH;
  if (i >= total) return;
  int    e   = (int)(i % KPATCH);
  size_t row = i / KPATCH;
  int b = (int)(row / NP), n = (int)(row % NP);
  int c = e >> 8, rr = e & 255, ph = rr >> 4, pw = rr & 15;
  int gy = n / 14, gx = n % 14;
  float v = px[(((size_t)b * 3 + c) * 224 + gy * 16 + ph) * 224 + gx * 16 + pw];
  out[i] = (_Float16)v;
}

// LayerNorm over HID=384: one block (128 threads) per row.
// Writes f16 row-major; optionally also f16 transposed [B,384,224] for attention.
template <bool TRANSP>
__global__ void __launch_bounds__(128)
ln_kernel(const float* __restrict__ x, const float* __restrict__ g,
          const float* __restrict__ bb, _Float16* __restrict__ out16,
          _Float16* __restrict__ outT) {
  int row  = blockIdx.x;                 // 0..50175
  int tid  = threadIdx.x;                // 0..127
  int lane = tid & 31, wid = tid >> 5;
  const float* xr = x + (size_t)row * HID;
  float v0 = xr[tid], v1 = xr[tid + 128], v2 = xr[tid + 256];

  __shared__ float red[4];
  float s = v0 + v1 + v2;
#pragma unroll
  for (int o = 16; o; o >>= 1) s += __shfl_down(s, o);
  if (!lane) red[wid] = s;
  __syncthreads();
  float mu = (red[0] + red[1] + red[2] + red[3]) * (1.f / HID);
  __syncthreads();

  float d0 = v0 - mu, d1 = v1 - mu, d2 = v2 - mu;
  float q = d0 * d0 + d1 * d1 + d2 * d2;
#pragma unroll
  for (int o = 16; o; o >>= 1) q += __shfl_down(q, o);
  if (!lane) red[wid] = q;
  __syncthreads();
  float var = (red[0] + red[1] + red[2] + red[3]) * (1.f / HID);
  float rs  = rsqrtf(var + 1e-5f);

  int   b   = row / NP;
  int   pos = row % NP;
  float d[3] = {d0, d1, d2};
#pragma unroll
  for (int j = 0; j < 3; ++j) {
    int   c = tid + 128 * j;
    float y = d[j] * rs * g[c] + bb[c];
    out16[(size_t)row * HID + c] = (_Float16)y;
    if (TRANSP)
      outT[((size_t)b * HID + c) * NPAD + pos] = (_Float16)y;
  }
}

// ---------------- generic WMMA GEMM (async-LDS double buffered) ----------------
// C[M,N] = A[M,K](f16,row-major) @ B, with B given as Bt[N,K] (f16, row-major)
// == PyTorch linear weight layout. M % 128 == 0, N % 64 == 0, K % 32 == 0.
template <bool BIAS, bool GELU_, bool RESID, bool POSEMB, bool OUTF32, bool OUTF16>
__global__ void __launch_bounds__(256)
gemm_kernel(const _Float16* __restrict__ A, const _Float16* __restrict__ Bt,
            float* __restrict__ Cf32, _Float16* __restrict__ Cf16,
            const float* __restrict__ bias, const float* __restrict__ resid,
            const float* __restrict__ posemb, int M, int N, int K) {
  __shared__ _Float16 sA[2][128 * 32];
  __shared__ _Float16 sB[2][64 * 32];

  int mBase = blockIdx.y * 128;
  int nBase = blockIdx.x * 64;
  int tid   = threadIdx.x;
  int lane  = tid & 31;
  int wave  = tid >> 5;
  int wm    = (wave & 3) * 32;   // wave's m offset in tile
  int wn    = (wave >> 2) * 32;  // wave's n offset in tile

  int ar = tid >> 1, ac = (tid & 1) * 16;  // A staging: 2x16B per thread
  int br = tid >> 2, bc = (tid & 3) * 8;   // B staging: 1x16B per thread

  const _Float16* gA = A + (size_t)(mBase + ar) * K + ac;
  const _Float16* gB = Bt + (size_t)(nBase + br) * K + bc;

  // Issue the 3 async b128 transfers for K-tile k0 into buffer `buf`.
  auto issue = [&](int buf, int k0) {
    _Float16* la = &sA[buf][ar * 32 + ac];
    _Float16* lb = &sB[buf][br * 32 + bc];
    async_b128(gA + k0, la);
    async_b128(gA + k0 + 8, la + 8);
    async_b128(gB + k0, lb);
  };

  v8f acc[2][2] = {};
  int nk = K / 32;
  issue(0, 0);
  for (int kt = 0; kt < nk; ++kt) {
    int buf = kt & 1;
    if (kt + 1 < nk) {
      issue(buf ^ 1, (kt + 1) * 32);
      wait_async<3>();  // previous tile done; prefetch (3 newest) in flight
    } else {
      wait_async<0>();
    }
    __syncthreads();

    v16h af[2], bf[2];
    af[0] = frag_load(&sA[buf][(wm + 0) * 32], 32);
    af[1] = frag_load(&sA[buf][(wm + 16) * 32], 32);
    bf[0] = frag_load(&sB[buf][(wn + 0) * 32], 32);
    bf[1] = frag_load(&sB[buf][(wn + 16) * 32], 32);
#pragma unroll
    for (int i = 0; i < 2; ++i)
#pragma unroll
      for (int j = 0; j < 2; ++j) acc[i][j] = wmma_f16(af[i], bf[j], acc[i][j]);
    __syncthreads();  // all waves done reading before this buffer is refilled
  }

  // epilogue; D layout: lane=(n, mhalf), vgpr v -> m = v + 8*(lane>>4)
  int r = lane & 15, h = lane >> 4;
#pragma unroll
  for (int i = 0; i < 2; ++i)
#pragma unroll
    for (int j = 0; j < 2; ++j)
#pragma unroll
      for (int v = 0; v < 8; ++v) {
        int    m   = mBase + wm + i * 16 + v + 8 * h;
        int    n   = nBase + wn + j * 16 + r;
        float  val = acc[i][j][v];
        if (BIAS)   val += bias[n];
        if (POSEMB) val += posemb[(size_t)(m % NP) * N + n];
        if (GELU_)  val = 0.5f * val * (1.0f + erff(val * 0.7071067811865476f));
        if (RESID)  val += resid[(size_t)m * N + n];
        if (OUTF32) Cf32[(size_t)m * N + n] = val;
        if (OUTF16) Cf16[(size_t)m * N + n] = (_Float16)val;
      }
}

// ---------------- fused attention ----------------
// One wave per (batch, 16-query tile). SA[b*196+row][0..31]=state, [32..63]=action.
// scores (13 WMMAs, K=32) -> masked softmax over 224 -> attn = W @ h via
// h16T[b][c][224] (zero-padded keys). Output attn16[B*196,384] f16.
__global__ void __launch_bounds__(32)
attn_kernel(const _Float16* __restrict__ SA, const _Float16* __restrict__ hT,
            _Float16* __restrict__ attn16) {
  __shared__ float    S[16 * NPAD];     // 14 KB raw scores
  __shared__ _Float16 Wl[16 * NPAD];    // 7 KB softmax weights (f16)

  int tile = blockIdx.x;   // 0..12 (13*16 = 208 >= 196)
  int b    = blockIdx.y;   // 0..255
  int mq   = tile * 16;
  int lane = threadIdx.x;
  int r    = lane & 15, h = lane >> 4;

  // A fragment: state rows (query). K = cols 0..31 of the 64-wide SA row.
  v16h aS = frag_load(SA + ((size_t)b * NP + mq) * 64, 64);

  const float invs = 0.17677669529663687f;  // 1/sqrt(32)
#pragma unroll 1
  for (int kt = 0; kt < 13; ++kt) {
    v16h aK  = frag_load(SA + ((size_t)b * NP + kt * 16) * 64 + 32, 64);
    v8f  acc = {};
    acc = wmma_f16(aS, aK, acc);
#pragma unroll
    for (int v = 0; v < 8; ++v)
      S[(v + 8 * h) * NPAD + kt * 16 + r] = acc[v] * invs;
  }
  // mask invalid keys 196..223
  for (int idx = lane; idx < 16 * (NPAD - NP); idx += 32) {
    int rr = idx / (NPAD - NP), k = NP + idx % (NPAD - NP);
    S[rr * NPAD + k] = -1e30f;
  }
  __syncthreads();

  // softmax: 2 lanes per row, 112 keys each, combine via shfl_xor(16)
  {
    float* Srow = S + r * NPAD + h * 112;
    float  mx   = -1e30f;
    for (int k = 0; k < 112; ++k) mx = fmaxf(mx, Srow[k]);
    mx = fmaxf(mx, __shfl_xor(mx, 16));
    float sum = 0.f;
    for (int k = 0; k < 112; ++k) sum += expf(Srow[k] - mx);
    sum += __shfl_xor(sum, 16);
    float     inv  = 1.f / sum;
    _Float16* Wrow = Wl + r * NPAD + h * 112;
    for (int k = 0; k < 112; ++k)
      Wrow[k] = (_Float16)(expf(Srow[k] - mx) * inv);
  }
  __syncthreads();

  // attn[16,384] = Wl[16,224] @ h[224,384]; B-operand from hT (N-major).
  size_t hbase = (size_t)b * HID * NPAD;
#pragma unroll 1
  for (int nc = 0; nc < 6; ++nc) {  // 6 chunks of 64 output cols
    v8f acc[4] = {};
#pragma unroll 1
    for (int kt = 0; kt < 7; ++kt) {
      v16h aW = frag_load(Wl + kt * 32, NPAD);
#pragma unroll
      for (int j = 0; j < 4; ++j) {
        int  c0 = nc * 64 + j * 16;
        v16h bH = frag_load(hT + hbase + (size_t)c0 * NPAD + kt * 32, NPAD);
        acc[j] = wmma_f16(aW, bH, acc[j]);
      }
    }
#pragma unroll
    for (int j = 0; j < 4; ++j)
#pragma unroll
      for (int v = 0; v < 8; ++v) {
        int m = v + 8 * h;
        int n = nc * 64 + j * 16 + r;
        if (mq + m < NP)
          attn16[((size_t)b * NP + mq + m) * HID + n] = (_Float16)acc[j][v];
      }
  }
}

// ---------------- workspace layout (bytes, all 256B aligned) ----------------
static const size_t OFF_X   = 0;
static const size_t SZ_X    = (size_t)MROWS * HID * 4;          //  77,070,336
static const size_t OFF_BIG = OFF_X + SZ_X;                     // P16/attn16/F16
static const size_t SZ_BIG  = (size_t)MROWS * FFND * 2;         // 154,140,672
static const size_t OFF_ACT = OFF_BIG + SZ_BIG;
static const size_t SZ_ACT  = (size_t)MROWS * HID * 2;
static const size_t OFF_HT  = OFF_ACT + SZ_ACT;
static const size_t SZ_HT   = (size_t)BDIM * HID * NPAD * 2;
static const size_t OFF_SA  = OFF_HT + SZ_HT;
static const size_t SZ_SA   = (size_t)(MROWS + 64) * 64 * 2;    // +slack rows
static const size_t OFF_WP  = OFF_SA + SZ_SA;                   // patchW16
static const size_t OFF_WSA = OFF_WP + (size_t)HID * KPATCH * 2;
static const size_t OFF_WO  = OFF_WSA + (size_t)3 * 64 * HID * 2;
static const size_t OFF_W1  = OFF_WO + (size_t)3 * HID * HID * 2;
static const size_t OFF_W2  = OFF_W1 + (size_t)3 * FFND * HID * 2;
static const size_t OFF_WPR = OFF_W2 + (size_t)3 * HID * FFND * 2;

extern "C" void kernel_launch(void* const* d_in, const int* in_sizes, int n_in,
                              void* d_out, int out_size, void* d_ws,
                              size_t ws_size, hipStream_t stream) {
  const float* pixel   = (const float*)d_in[0];
  const float* patch_W = (const float*)d_in[1];
  const float* pos_emb = (const float*)d_in[2];
  const float* ln1_g   = (const float*)d_in[3];
  const float* ln1_b   = (const float*)d_in[4];
  const float* Ws      = (const float*)d_in[5];
  const float* Wa      = (const float*)d_in[6];
  const float* Wo      = (const float*)d_in[7];
  const float* bo      = (const float*)d_in[8];
  const float* ln2_g   = (const float*)d_in[9];
  const float* ln2_b   = (const float*)d_in[10];
  const float* W1      = (const float*)d_in[11];
  const float* b1      = (const float*)d_in[12];
  const float* W2      = (const float*)d_in[13];
  const float* b2      = (const float*)d_in[14];
  const float* lnf_g   = (const float*)d_in[15];
  const float* lnf_b   = (const float*)d_in[16];
  const float* proj_W  = (const float*)d_in[17];
  float*       out     = (float*)d_out;
  char*        ws      = (char*)d_ws;

  float*    x     = (float*)(ws + OFF_X);
  _Float16* big   = (_Float16*)(ws + OFF_BIG);  // P16 -> attn16 -> F16 (reused)
  _Float16* act16 = (_Float16*)(ws + OFF_ACT);
  _Float16* hT    = (_Float16*)(ws + OFF_HT);
  _Float16* SA16  = (_Float16*)(ws + OFF_SA);
  _Float16* wP    = (_Float16*)(ws + OFF_WP);
  _Float16* wSA   = (_Float16*)(ws + OFF_WSA);
  _Float16* wO    = (_Float16*)(ws + OFF_WO);
  _Float16* w1f   = (_Float16*)(ws + OFF_W1);
  _Float16* w2f   = (_Float16*)(ws + OFF_W2);
  _Float16* wPR   = (_Float16*)(ws + OFF_WPR);

  // --- weight conversion (tiny; PyTorch [N,K] layout == Bt layout we need) ---
  f32_to_f16_kernel<<<1024, 256, 0, stream>>>(patch_W, wP, (long)HID * KPATCH);
  f32_to_f16_kernel<<<1024, 256, 0, stream>>>(Wo, wO, (long)3 * HID * HID);
  f32_to_f16_kernel<<<1024, 256, 0, stream>>>(W1, w1f, (long)3 * FFND * HID);
  f32_to_f16_kernel<<<1024, 256, 0, stream>>>(W2, w2f, (long)3 * HID * FFND);
  f32_to_f16_kernel<<<1024, 256, 0, stream>>>(proj_W, wPR, (long)DOUTD * HID);
  build_wsa_kernel<<<(3 * 64 * HID + 255) / 256, 256, 0, stream>>>(Ws, Wa, wSA);
  zero_pad_hT_kernel<<<(BDIM * HID * (NPAD - NP) + 255) / 256, 256, 0, stream>>>(hT);

  // --- patch embed: im2col + GEMM (+pos_emb) ---
  {
    size_t total = (size_t)MROWS * KPATCH;
    im2col_kernel<<<(unsigned)((total + 255) / 256), 256, 0, stream>>>(pixel, big);
    gemm_kernel<false, false, false, true, true, false>
        <<<dim3(HID / 64, MROWS / 128), 256, 0, stream>>>(
            big, wP, x, nullptr, nullptr, nullptr, pos_emb, MROWS, HID, KPATCH);
  }

  for (int i = 0; i < 3; ++i) {
    // LN1 -> act16 (GEMM-A) + hT (transposed, padded)
    ln_kernel<true><<<MROWS, 128, 0, stream>>>(x, ln1_g + i * HID,
                                               ln1_b + i * HID, act16, hT);
    // [state|action] = h @ [Ws|Wa]^T  (N=64), f16 out
    gemm_kernel<false, false, false, false, false, true>
        <<<dim3(1, MROWS / 128), 256, 0, stream>>>(
            act16, wSA + (size_t)i * 64 * HID, nullptr, SA16, nullptr, nullptr,
            nullptr, MROWS, 64, HID);
    // fused scores/softmax/weighted-sum -> attn16 (aliases `big`)
    attn_kernel<<<dim3(13, BDIM), 32, 0, stream>>>(SA16, hT, big);
    // x += attn @ Wo^T + bo
    gemm_kernel<true, false, true, false, true, false>
        <<<dim3(HID / 64, MROWS / 128), 256, 0, stream>>>(
            big, wO + (size_t)i * HID * HID, x, nullptr, bo + i * HID, x,
            nullptr, MROWS, HID, HID);
    // LN2 -> act16
    ln_kernel<false><<<MROWS, 128, 0, stream>>>(x, ln2_g + i * HID,
                                                ln2_b + i * HID, act16, nullptr);
    // F = gelu(h2 @ W1^T + b1), f16 out (aliases `big`; attn16 is dead)
    gemm_kernel<true, true, false, false, false, true>
        <<<dim3(FFND / 64, MROWS / 128), 256, 0, stream>>>(
            act16, w1f + (size_t)i * FFND * HID, nullptr, big, b1 + i * FFND,
            nullptr, nullptr, MROWS, FFND, HID);
    // x += F @ W2^T + b2
    gemm_kernel<true, false, true, false, true, false>
        <<<dim3(HID / 64, MROWS / 128), 256, 0, stream>>>(
            big, w2f + (size_t)i * HID * FFND, x, nullptr, b2 + i * HID, x,
            nullptr, MROWS, HID, FFND);
  }

  // final LN + projection
  ln_kernel<false><<<MROWS, 128, 0, stream>>>(x, lnf_g, lnf_b, act16, nullptr);
  gemm_kernel<false, false, false, false, true, false>
      <<<dim3(DOUTD / 64, MROWS / 128), 256, 0, stream>>>(
          act16, wPR, out, nullptr, nullptr, nullptr, nullptr, MROWS, DOUTD, HID);
}
